// BestRQModel_34110630265196
// MI455X (gfx1250) — matrix operand order repaired and checked by reading, not compiled, not run on hardware
//
#include <hip/hip_runtime.h>

// ---------------------------------------------------------------------------
// BestRQ forward pass for MI455X (gfx1250, wave32, WMMA 16x16x32 f16).
// D=512, E=512, V=8192, NC=1, L=6, H=8, F=2048, HD=64, B=8, T=1024.
// All GEMM operands pre-converted to fp16 (weights transposed to N-major) so
// WMMA fragments are contiguous 16B runs. GEMM tiles stream through LDS with
// double-buffered global_load_async_to_lds_b128 + s_wait_asynccnt. Attention
// uses a transposed V layout and f16 probability tile so every fragment is a
// pair of 128-bit vector loads.
// ---------------------------------------------------------------------------

typedef __attribute__((ext_vector_type(16))) _Float16 v16h;
typedef __attribute__((ext_vector_type(8)))  _Float16 v8h;
typedef __attribute__((ext_vector_type(8)))  float    v8f;

#define WMMA_F16(a, b, c) \
  __builtin_amdgcn_wmma_f32_16x16x32_f16(false, (a), false, (b), (short)0, (c), false, false)

__device__ __forceinline__ v16h cat8(v8h x, v8h y) {
  return __builtin_shufflevector(x, y, 0, 1, 2, 3, 4, 5, 6, 7,
                                       8, 9, 10, 11, 12, 13, 14, 15);
}
// A fragment (16x32 MxK, 16-bit): halves 0..7 = K 8*hi..8*hi+7,
// halves 8..15 = K 16+8*hi..23+8*hi  -> two contiguous v8h chunks.
__device__ __forceinline__ v16h frag_a(const _Float16* p, int hi) {
  return cat8(*(const v8h*)(p + 8 * hi), *(const v8h*)(p + 16 + 8 * hi));
}
// B fragment (32x16 KxN, 16-bit), operand stored N-major (B^T row):
// halves 0..7 = K 8*hi..8*hi+7, halves 8..15 = K 16+8*hi..23+8*hi (same as A).
__device__ __forceinline__ v16h frag_b(const _Float16* p, int hi) {
  return cat8(*(const v8h*)(p + 8 * hi), *(const v8h*)(p + 16 + 8 * hi));
}
__device__ __forceinline__ unsigned lds_off(const void* p) {
  return (unsigned)(unsigned long long)p;   // low 32 bits of generic LDS addr
}
__device__ __forceinline__ void async_copy_b128(unsigned lds, const _Float16* g) {
  asm volatile("global_load_async_to_lds_b128 %0, %1, off"
               :: "v"(lds), "v"(g) : "memory");
}
__device__ __forceinline__ void wait_async0() {
  asm volatile("s_wait_asynccnt 0x0" ::: "memory");
}
__device__ __forceinline__ void wait_async4() {  // prev tile's 4 copies done
  asm volatile("s_wait_asynccnt 0x4" ::: "memory");
}

// ----------------------------- fp32 -> fp16 prep ----------------------------
__global__ void cvt_kernel(const float* __restrict__ in, _Float16* __restrict__ out) {
  int idx = blockIdx.x * 256 + threadIdx.x;
  out[idx] = (_Float16)in[idx];
}
// out[n*K + k] = (f16) in[k*N + n]   (weight transpose to N-major)
__global__ void cvtT_kernel(const float* __restrict__ in, _Float16* __restrict__ out,
                            int K, int N) {
  int idx = blockIdx.x * 256 + threadIdx.x;
  int n = idx / K, k = idx - n * K;
  out[(size_t)n * K + k] = (_Float16)in[(size_t)k * N + n];
}
// V section of qkv -> VT[b][h][hd][t] (f16), coalesced over t.
__global__ void vT_kernel(const _Float16* __restrict__ qkv, _Float16* __restrict__ VT) {
  int idx = blockIdx.x * 256 + threadIdx.x;   // B*H*HD*T = 4194304
  int t  = idx & 1023;
  int hd = (idx >> 10) & 63;
  int h  = (idx >> 16) & 7;
  int b  = idx >> 19;
  VT[idx] = qkv[((size_t)(b * 1024 + t)) * 1536 + 1024 + h * 64 + hd];
}

// ------------------------------- generic GEMM -------------------------------
// C[M,N] = A[M,K] @ B[K,N] (+ epilogue); A f16 row-major, Bt f16 N-major.
// 128 threads = 4 waves, 64x64 tile; double-buffered async LDS tiles.
enum { EPI_NONE = 0, EPI_PE = 1, EPI_RES = 2, EPI_RELU = 3 };

template <int EPI, typename OutT>
__global__ __launch_bounds__(128)
void gemm_kernel(const _Float16* __restrict__ A, const _Float16* __restrict__ Bt,
                 OutT* __restrict__ C, const float* __restrict__ Res,
                 int M, int N, int K) {
  __shared__ _Float16 As[2][64][32];   // [buf][m][k]
  __shared__ _Float16 Bs[2][64][32];   // [buf][n][k]  (B^T tile)
  const int tid = threadIdx.x, lane = tid & 31, wave = tid >> 5;
  const int hi = lane >> 4, lm = lane & 15;
  const int m0 = blockIdx.y * 64, n0 = blockIdx.x * 64;

  // Two 16B chunks per tile per thread: 256 chunks cover 64 rows x 32 halves.
  const int e0 = tid * 2, e1 = e0 + 1;
  const int r0 = e0 >> 2, p0 = (e0 & 3) * 8;
  const int r1 = e1 >> 2, p1 = (e1 & 3) * 8;
  const unsigned lA0[2] = {lds_off(&As[0][r0][p0]), lds_off(&As[1][r0][p0])};
  const unsigned lA1[2] = {lds_off(&As[0][r1][p1]), lds_off(&As[1][r1][p1])};
  const unsigned lB0[2] = {lds_off(&Bs[0][r0][p0]), lds_off(&Bs[1][r0][p0])};
  const unsigned lB1[2] = {lds_off(&Bs[0][r1][p1]), lds_off(&Bs[1][r1][p1])};
  const _Float16* ga0 = A + (size_t)(m0 + r0) * K + p0;
  const _Float16* ga1 = A + (size_t)(m0 + r1) * K + p1;
  const _Float16* gb0 = Bt + (size_t)(n0 + r0) * K + p0;
  const _Float16* gb1 = Bt + (size_t)(n0 + r1) * K + p1;

  // Prologue: stream tile 0 into buffer 0.
  async_copy_b128(lA0[0], ga0);
  async_copy_b128(lA1[0], ga1);
  async_copy_b128(lB0[0], gb0);
  async_copy_b128(lB1[0], gb1);

  v8f c[4] = {};
  int p = 0;
  for (int ks = 0; ks < K; ks += 32) {
    const bool last = (ks + 32 >= K);
    if (!last) {                       // stream next tile into other buffer
      const int kn = ks + 32;
      async_copy_b128(lA0[p ^ 1], ga0 + kn);
      async_copy_b128(lA1[p ^ 1], ga1 + kn);
      async_copy_b128(lB0[p ^ 1], gb0 + kn);
      async_copy_b128(lB1[p ^ 1], gb1 + kn);
      wait_async4();                   // in-order retire => current tile ready
    } else {
      wait_async0();
    }
    __syncthreads();
    const v16h a = frag_a(&As[p][wave * 16 + lm][0], hi);
#pragma unroll
    for (int jt = 0; jt < 4; ++jt) {
      const v16h b = frag_b(&Bs[p][jt * 16 + lm][0], hi);
      c[jt] = WMMA_F16(a, b, c[jt]);
    }
    __syncthreads();
    p ^= 1;
  }

#pragma unroll
  for (int jt = 0; jt < 4; ++jt) {
#pragma unroll
    for (int r8 = 0; r8 < 8; ++r8) {
      int row = m0 + wave * 16 + r8 + 8 * hi;   // C: M = r8 + 8*hi
      int col = n0 + jt * 16 + lm;              //    N = lane & 15
      float v = c[jt][r8];
      if (EPI == EPI_PE) {  // sinusoidal PE, t = row % 1024
        int t  = row & 1023;
        int ie = col & ~1;
        float ang = (float)t * __expf(-(float)ie * (9.210340371976184f / 512.0f));
        v += (col & 1) ? __cosf(ang) : __sinf(ang);
      }
      if (EPI == EPI_RES)  v += Res[(size_t)row * N + col];
      if (EPI == EPI_RELU) v = fmaxf(v, 0.0f);
      C[(size_t)row * N + col] = (OutT)v;
    }
  }
}

// ---------------------- layernorm (fp32 in, fp16 out) -----------------------
__global__ __launch_bounds__(128)
void ln_kernel(const float* __restrict__ X, const float* __restrict__ s,
               const float* __restrict__ b, _Float16* __restrict__ Y) {
  __shared__ float red[128];
  const int row = blockIdx.x;
  const int tid = threadIdx.x;
  const float* xr = X + (size_t)row * 512;
  float v[4], sum = 0.0f;
#pragma unroll
  for (int i = 0; i < 4; ++i) { v[i] = xr[tid + i * 128]; sum += v[i]; }
  red[tid] = sum; __syncthreads();
  for (int off = 64; off > 0; off >>= 1) {
    if (tid < off) red[tid] += red[tid + off];
    __syncthreads();
  }
  const float mean = red[0] * (1.0f / 512.0f);
  __syncthreads();
  float vs = 0.0f;
#pragma unroll
  for (int i = 0; i < 4; ++i) { float d = v[i] - mean; vs += d * d; }
  red[tid] = vs; __syncthreads();
  for (int off = 64; off > 0; off >>= 1) {
    if (tid < off) red[tid] += red[tid + off];
    __syncthreads();
  }
  const float inv = rsqrtf(red[0] * (1.0f / 512.0f) + 1e-5f);
#pragma unroll
  for (int i = 0; i < 4; ++i) {
    int cc = tid + i * 128;
    Y[(size_t)row * 512 + cc] = (_Float16)((v[i] - mean) * inv * s[cc] + b[cc]);
  }
}

// ----------------------------- span mask write ------------------------------
__global__ void mask_kernel(float* __restrict__ x, const unsigned char* __restrict__ mask,
                            const float* __restrict__ memb) {
  int idx = blockIdx.x * 256 + threadIdx.x;   // B*T*D = 4194304
  int row = idx >> 9, col = idx & 511;
  if (mask[row]) x[idx] = memb[col];
}

// ---------------------- codebook column norms |e_v|^2 -----------------------
__global__ void enorm_kernel(const float* __restrict__ emb, float* __restrict__ enorm) {
  int v = blockIdx.x * 256 + threadIdx.x;     // V = 8192
  float s = 0.0f;
  for (int e = 0; e < 512; ++e) { float t = emb[(size_t)e * 8192 + v]; s += t * t; }
  enorm[v] = s;
}

// ------------- fused z@E GEMM + argmin over V (one wave, 16 rows) -----------
__global__ __launch_bounds__(32)
void nearest_kernel(const _Float16* __restrict__ Z, const _Float16* __restrict__ EmbT,
                    const float* __restrict__ enorm, int* __restrict__ tgt) {
  __shared__ _Float16 zs[16][512];
  const int lane = threadIdx.x, hi = lane >> 4, lm = lane & 15;
  const int row0 = blockIdx.x * 16;
  for (int i = 0; i < 32; ++i) {              // 1024 v8h chunks, 32 per lane
    int e = lane * 32 + i;
    int r = e >> 6, cc = (e & 63) * 8;
    *(v8h*)&zs[r][cc] = *(const v8h*)&Z[(size_t)(row0 + r) * 512 + cc];
  }
  __syncthreads();

  float bv[8]; int bi[8];
#pragma unroll
  for (int r8 = 0; r8 < 8; ++r8) { bv[r8] = -3.4e38f; bi[r8] = 0; }

  for (int nb = 0; nb < 512; ++nb) {          // maximize 2*z.e - |e|^2
    v8f c = {};
    const int col = nb * 16 + lm;
    const _Float16* ecol = EmbT + (size_t)col * 512;
    for (int ks = 0; ks < 16; ++ks) {
      const v16h a  = frag_a(&zs[lm][ks * 32], hi);
      const v16h bf = frag_b(ecol + ks * 32, hi);
      c = WMMA_F16(a, bf, c);
    }
    const float en = enorm[col];
#pragma unroll
    for (int r8 = 0; r8 < 8; ++r8) {
      float val = 2.0f * c[r8] - en;
      if (val > bv[r8]) { bv[r8] = val; bi[r8] = col; }
    }
  }
#pragma unroll
  for (int off = 1; off <= 8; off <<= 1) {
#pragma unroll
    for (int r8 = 0; r8 < 8; ++r8) {
      float ov = __shfl_xor(bv[r8], off, 32);
      int   oi = __shfl_xor(bi[r8], off, 32);
      if (ov > bv[r8] || (ov == bv[r8] && oi < bi[r8])) { bv[r8] = ov; bi[r8] = oi; }
    }
  }
  if (lm == 0)
#pragma unroll
    for (int r8 = 0; r8 < 8; ++r8) tgt[row0 + r8 + 8 * hi] = bi[r8];
}

// --------------------- attention: one wave per 16 q-rows --------------------
__global__ __launch_bounds__(32)
void attn_kernel(const _Float16* __restrict__ QKV, const _Float16* __restrict__ VT,
                 const int* __restrict__ lens, _Float16* __restrict__ O) {
  __shared__ float    S[16][1024];            // 64 KB raw scores (fp32)
  __shared__ _Float16 Sp[16][1024];           // 32 KB probabilities (f16)
  const int lane = threadIdx.x, hi = lane >> 4, lm = lane & 15;
  const int qt = blockIdx.x & 63;
  const int h  = (blockIdx.x >> 6) & 7;
  const int b  = blockIdx.x >> 9;
  const int len = lens[b];
  const int t0  = qt * 16;
  const size_t ld = 1536;
  const _Float16* base = QKV + (size_t)b * 1024 * ld;

  // Q fragments (K-dim HD=64, split into two WMMAs); scale 1/sqrt(HD) folded.
  const _Float16* qr = base + (size_t)(t0 + lm) * ld + h * 64;
  v16h a0 = frag_a(qr, hi);
  v16h a1 = frag_a(qr + 32, hi);
  a0 *= (_Float16)0.125f;
  a1 *= (_Float16)0.125f;

  // S = Q @ K^T + bias  (K rows contiguous -> frag_b vector loads)
  for (int kt = 0; kt < 64; ++kt) {
    const _Float16* kp = base + (size_t)(kt * 16 + lm) * ld + 512 + h * 64;
    v8f c = {};
    c = WMMA_F16(a0, frag_b(kp, hi), c);
    c = WMMA_F16(a1, frag_b(kp + 32, hi), c);
    const int col = kt * 16 + lm;
    const float bias = (col < len) ? 0.0f : -1e9f;
#pragma unroll
    for (int r8 = 0; r8 < 8; ++r8) S[r8 + 8 * hi][col] = c[r8] + bias;
  }
  __syncthreads();

  // softmax: lane pair (lm, lm+16) handles row lm, 512 columns each;
  // normalized probabilities stored as f16 for vectorized P fragments.
  {
    float* sr = &S[lm][hi * 512];
    _Float16* pr = &Sp[lm][hi * 512];
    float m = -3.4e38f;
    for (int i = 0; i < 512; ++i) m = fmaxf(m, sr[i]);
    m = fmaxf(m, __shfl_xor(m, 16, 32));
    float s = 0.0f;
    for (int i = 0; i < 512; ++i) { float e = __expf(sr[i] - m); pr[i] = (_Float16)e; s += e; }
    s += __shfl_xor(s, 16, 32);
    const float inv = 1.0f / s;
    for (int i = 0; i < 512; ++i) pr[i] = (_Float16)((float)pr[i] * inv);
  }
  __syncthreads();

  // O = P @ V  (P from f16 LDS tile, V from transposed [b][h][hd][t] layout)
  const _Float16* vbase = VT + (size_t)(b * 8 + h) * 64 * 1024;
  v8f o[4] = {};
  for (int kt = 0; kt < 32; ++kt) {
    const v16h a = frag_a(&Sp[lm][kt * 32], hi);
#pragma unroll
    for (int jt = 0; jt < 4; ++jt) {
      const _Float16* vb = vbase + (size_t)(jt * 16 + lm) * 1024 + kt * 32;
      o[jt] = WMMA_F16(a, frag_b(vb, hi), o[jt]);
    }
  }
#pragma unroll
  for (int jt = 0; jt < 4; ++jt)
#pragma unroll
    for (int r8 = 0; r8 < 8; ++r8) {
      int row = t0 + r8 + 8 * hi;
      O[((size_t)b * 1024 + row) * 512 + h * 64 + jt * 16 + lm] = (_Float16)o[jt][r8];
    }
}

// ------- fused logits GEMM + online log-sum-exp + target gather/CE ----------
__global__ __launch_bounds__(32)
void loss_kernel(const _Float16* __restrict__ Hout, const _Float16* __restrict__ WvT,
                 const int* __restrict__ tgt, const int* __restrict__ lens,
                 const unsigned char* __restrict__ mask, float* __restrict__ ent_out) {
  __shared__ _Float16 hs[16][512];
  const int lane = threadIdx.x, hi = lane >> 4, lm = lane & 15;
  const int row0 = blockIdx.x * 16;
  for (int i = 0; i < 32; ++i) {
    int e = lane * 32 + i;
    int r = e >> 6, cc = (e & 63) * 8;
    *(v8h*)&hs[r][cc] = *(const v8h*)&Hout[(size_t)(row0 + r) * 512 + cc];
  }
  __syncthreads();

  float mx[8], ss[8], tl[8]; int tg[8];
#pragma unroll
  for (int r8 = 0; r8 < 8; ++r8) {
    mx[r8] = -3.4e38f; ss[r8] = 0.0f; tl[r8] = -3.4e38f;
    tg[r8] = tgt[row0 + r8 + 8 * hi];
  }

  for (int nb = 0; nb < 512; ++nb) {
    v8f c = {};
    const int col = nb * 16 + lm;
    const _Float16* wcol = WvT + (size_t)col * 512;
    for (int ks = 0; ks < 16; ++ks) {
      const v16h a  = frag_a(&hs[lm][ks * 32], hi);
      const v16h bf = frag_b(wcol + ks * 32, hi);
      c = WMMA_F16(a, bf, c);
    }
#pragma unroll
    for (int r8 = 0; r8 < 8; ++r8) {
      float val = c[r8];
      if (val > mx[r8]) { ss[r8] = ss[r8] * __expf(mx[r8] - val) + 1.0f; mx[r8] = val; }
      else              { ss[r8] += __expf(val - mx[r8]); }
      if (col == tg[r8]) tl[r8] = val;
    }
  }
#pragma unroll
  for (int off = 1; off <= 8; off <<= 1) {
#pragma unroll
    for (int r8 = 0; r8 < 8; ++r8) {
      float om = __shfl_xor(mx[r8], off, 32);
      float os = __shfl_xor(ss[r8], off, 32);
      float nm = fmaxf(mx[r8], om);
      ss[r8] = ss[r8] * __expf(mx[r8] - nm) + os * __expf(om - nm);
      mx[r8] = nm;
      tl[r8] = fmaxf(tl[r8], __shfl_xor(tl[r8], off, 32));
    }
  }
  if (lm == 0)
#pragma unroll
    for (int r8 = 0; r8 < 8; ++r8) {
      int row = row0 + r8 + 8 * hi;
      int bb = row >> 10, t = row & 1023;
      float w = (t < lens[bb] && mask[row]) ? 1.0f : 0.0f;
      ent_out[row] = w * ((mx[r8] + __logf(ss[r8])) - tl[r8]);
    }
}

// ------------------ deterministic final reduction to scalar -----------------
__global__ __launch_bounds__(256)
void final_kernel(const float* __restrict__ ent, const int* __restrict__ lens,
                  const unsigned char* __restrict__ mask, float* __restrict__ out) {
  __shared__ float rs[256];
  __shared__ float rc[256];
  const int tid = threadIdx.x;
  float s = 0.0f, cnt = 0.0f;
  for (int i = tid; i < 8192; i += 256) {
    s += ent[i];
    int bb = i >> 10, t = i & 1023;
    cnt += (t < lens[bb] && mask[i]) ? 1.0f : 0.0f;
  }
  rs[tid] = s; rc[tid] = cnt; __syncthreads();
  for (int off = 128; off > 0; off >>= 1) {
    if (tid < off) { rs[tid] += rs[tid + off]; rc[tid] += rc[tid + off]; }
    __syncthreads();
  }
  if (tid == 0) out[0] = rs[0] / fmaxf(rc[0], 1.0f);   // NC == 1
}

// ---------------------------------------------------------------------------
extern "C" void kernel_launch(void* const* d_in, const int* in_sizes, int n_in,
                              void* d_out, int out_size, void* d_ws, size_t ws_size,
                              hipStream_t stream) {
  const float*         xs     = (const float*)d_in[0];
  const int*           lens   = (const int*)d_in[1];
  const unsigned char* maskb  = (const unsigned char*)d_in[2];
  const float*         Wemb   = (const float*)d_in[3];
  const float*         ln1s   = (const float*)d_in[4];
  const float*         ln1b   = (const float*)d_in[5];
  const float*         Wqkv   = (const float*)d_in[6];
  const float*         Wo     = (const float*)d_in[7];
  const float*         ln2s   = (const float*)d_in[8];
  const float*         ln2b   = (const float*)d_in[9];
  const float*         W1     = (const float*)d_in[10];
  const float*         W2     = (const float*)d_in[11];
  const float*         ans    = (const float*)d_in[12];
  const float*         anb    = (const float*)d_in[13];
  const float*         ilns   = (const float*)d_in[14];
  const float*         ilnb   = (const float*)d_in[15];
  const float*         memb   = (const float*)d_in[16];
  const float*         Wtop   = (const float*)d_in[17];  // [512, 8192]
  const float*         proj   = (const float*)d_in[18];
  const float*         emb    = (const float*)d_in[19];  // [512, 8192]

  // ---- workspace carve-out (256B aligned) ----
  char* wsb = (char*)d_ws;
  size_t off = 0;
  auto carve = [&](size_t bytes) -> void* {
    void* p = wsb + off; off += (bytes + 255) & ~(size_t)255; return p;
  };
  float*     x     = (float*)carve(4194304ull * 4);       // residual stream fp32
  _Float16*  hbuf  = (_Float16*)carve(4194304ull * 2);    // LN outputs (f16)
  _Float16*  big   = (_Float16*)carve(16777216ull * 2);   // qkv|attn / ffn1 / z
  _Float16*  vt    = (_Float16*)carve(4194304ull * 2);    // V transposed (f16)
  float*     enrm  = (float*)carve(8192ull * 4);
  int*       tgt   = (int*)carve(8192ull * 4);
  float*     entb  = (float*)carve(8192ull * 4);
  _Float16*  xs16  = (_Float16*)carve(4194304ull * 2);
  _Float16*  WembT = (_Float16*)carve(262144ull * 2);
  _Float16*  projT = (_Float16*)carve(262144ull * 2);
  _Float16*  WqkvT = (_Float16*)carve(4718592ull * 2);
  _Float16*  WoT   = (_Float16*)carve(1572864ull * 2);
  _Float16*  W1T   = (_Float16*)carve(6291456ull * 2);
  _Float16*  W2T   = (_Float16*)carve(6291456ull * 2);
  _Float16*  WtopT = (_Float16*)carve(4194304ull * 2);
  _Float16*  embT  = (_Float16*)carve(4194304ull * 2);

  _Float16* qkv  = big;                 // 12,582,912 halves
  _Float16* attn = big + 12582912;      //  4,194,304 halves
  _Float16* ffn1 = big;                 // 16,777,216 halves (post-attn phase)
  _Float16* zbuf = big;                 //  4,194,304 halves (pre-encoder)

  const dim3 blk(128);

  // 0) fp16 operand prep (once per launch)
  cvt_kernel<<<16384, 256, 0, stream>>>(xs, xs16);
  cvtT_kernel<<<1024, 256, 0, stream>>>(Wemb, WembT, 512, 512);
  cvtT_kernel<<<1024, 256, 0, stream>>>(proj, projT, 512, 512);
  for (int l = 0; l < 6; ++l) {
    cvtT_kernel<<<3072, 256, 0, stream>>>(Wqkv + (size_t)l * 786432, WqkvT + (size_t)l * 786432, 512, 1536);
    cvtT_kernel<<<1024, 256, 0, stream>>>(Wo + (size_t)l * 262144, WoT + (size_t)l * 262144, 512, 512);
    cvtT_kernel<<<4096, 256, 0, stream>>>(W1 + (size_t)l * 1048576, W1T + (size_t)l * 1048576, 512, 2048);
    cvtT_kernel<<<4096, 256, 0, stream>>>(W2 + (size_t)l * 1048576, W2T + (size_t)l * 1048576, 2048, 512);
  }
  cvtT_kernel<<<16384, 256, 0, stream>>>(Wtop, WtopT, 512, 8192);
  cvtT_kernel<<<16384, 256, 0, stream>>>(emb, embT, 512, 8192);

  // 1) x = xs @ W_embed + PE
  gemm_kernel<EPI_PE, float><<<dim3(8, 128), blk, 0, stream>>>(
      xs16, WembT, x, nullptr, 8192, 512, 512);

  // 2) target ids: z = LN(x) @ projection; argmin_v |z - e_v|^2
  ln_kernel<<<8192, 128, 0, stream>>>(x, ilns, ilnb, hbuf);
  gemm_kernel<EPI_NONE, _Float16><<<dim3(8, 128), blk, 0, stream>>>(
      hbuf, projT, zbuf, nullptr, 8192, 512, 512);
  enorm_kernel<<<32, 256, 0, stream>>>(emb, enrm);
  nearest_kernel<<<512, 32, 0, stream>>>(zbuf, embT, enrm, tgt);

  // 3) masked_xs = where(mask, mask_emb, x)   (in place)
  mask_kernel<<<16384, 256, 0, stream>>>(x, maskb, memb);

  // 4) 6 encoder layers
  for (int l = 0; l < 6; ++l) {
    ln_kernel<<<8192, 128, 0, stream>>>(x, ln1s + l * 512, ln1b + l * 512, hbuf);
    gemm_kernel<EPI_NONE, _Float16><<<dim3(24, 128), blk, 0, stream>>>(
        hbuf, WqkvT + (size_t)l * 786432, qkv, nullptr, 8192, 1536, 512);
    vT_kernel<<<16384, 256, 0, stream>>>(qkv, vt);
    attn_kernel<<<4096, 32, 0, stream>>>(qkv, vt, lens, attn);
    gemm_kernel<EPI_RES, float><<<dim3(8, 128), blk, 0, stream>>>(
        attn, WoT + (size_t)l * 262144, x, x, 8192, 512, 512);
    ln_kernel<<<8192, 128, 0, stream>>>(x, ln2s + l * 512, ln2b + l * 512, hbuf);
    gemm_kernel<EPI_RELU, _Float16><<<dim3(32, 128), blk, 0, stream>>>(
        hbuf, W1T + (size_t)l * 1048576, ffn1, nullptr, 8192, 2048, 512);
    gemm_kernel<EPI_RES, float><<<dim3(8, 128), blk, 0, stream>>>(
        ffn1, W2T + (size_t)l * 1048576, x, x, 8192, 512, 2048);
  }

  // 5) final LN, logits + masked cross-entropy
  ln_kernel<<<8192, 128, 0, stream>>>(x, ans, anb, hbuf);
  loss_kernel<<<512, 32, 0, stream>>>(hbuf, WtopT, tgt, lens, maskb, entb);
  final_kernel<<<1, 256, 0, stream>>>(entb, lens, maskb, (float*)d_out);
}
